// ChebyshevConvolution_81140522156081
// MI455X (gfx1250) — matrix operand order repaired and checked by reading, not compiled, not run on hardware
//
#include <hip/hip_runtime.h>
#include <hip/hip_bf16.h>
#include <stdint.h>

#define N_NODES 200000
#define N_EDGES 3200000
#define F_IN    165
#define HIDDEN  128
#define KTOT    495          // 3 * 165 (virtual K of fused layer-1 GEMM)
#define KPAD    512
#define NCHUNK  16           // KPAD / 32

typedef __attribute__((ext_vector_type(16))) _Float16 v16h;
typedef __attribute__((ext_vector_type(8)))  _Float16 v8h;
typedef __attribute__((ext_vector_type(8)))  float    v8f;

// ---------------- graph preprocessing ----------------

__global__ void k_degree(const long long* __restrict__ ei, float* __restrict__ deg) {
  int e = blockIdx.x * 256 + threadIdx.x;
  if (e >= N_EDGES) return;
  atomicAdd(&deg[ei[e]], 1.0f);
}

__global__ void k_dis(const float* __restrict__ deg, float* __restrict__ dis) {
  int n = blockIdx.x * 256 + threadIdx.x;
  if (n >= N_NODES) return;
  float d = deg[n];
  dis[n] = d > 0.0f ? rsqrtf(d) : 0.0f;
}

__global__ void k_norm(const long long* __restrict__ ei, const float* __restrict__ dis,
                       float* __restrict__ nrm) {
  int e = blockIdx.x * 256 + threadIdx.x;
  if (e >= N_EDGES) return;
  long long r = ei[e], c = ei[(size_t)N_EDGES + e];
  nrm[e] = -dis[r] * dis[c];
}

__global__ void k_neg(const float* __restrict__ x, float* __restrict__ y, int count) {
  int i = blockIdx.x * 256 + threadIdx.x;
  if (i < count) y[i] = -x[i];
}

// out[row,:] += scale * norm[e] * in[col,:]   (165-wide, feature-chunked)
__global__ void k_prop165(const long long* __restrict__ ei, const float* __restrict__ nrm,
                          const float* __restrict__ in, float* __restrict__ out, float scale) {
  int e = blockIdx.x * 256 + threadIdx.x;
  if (e >= N_EDGES) return;
  int f0 = blockIdx.y * 28;
  int f1 = f0 + 28 < F_IN ? f0 + 28 : F_IN;
  long long r = ei[e], c = ei[(size_t)N_EDGES + e];
  float s = scale * nrm[e];
  const float* src = in + (size_t)c * F_IN;
  float* dst = out + (size_t)r * F_IN;
  for (int f = f0; f < f1; ++f) atomicAdd(&dst[f], s * src[f]);
}

// 2-wide propagation (layer 2, after projection)
__global__ void k_prop2(const long long* __restrict__ ei, const float* __restrict__ nrm,
                        const float* __restrict__ in, float* __restrict__ out) {
  int e = blockIdx.x * 256 + threadIdx.x;
  if (e >= N_EDGES) return;
  long long r = ei[e], c = ei[(size_t)N_EDGES + e];
  float s = nrm[e];
  atomicAdd(&out[r * 2 + 0], s * in[c * 2 + 0]);
  atomicAdd(&out[r * 2 + 1], s * in[c * 2 + 1]);
}

// ---------------- WMMA GEMM (layer 1) ----------------
// Pre-pack W1 (495x128 fp32, zero-padded to 512) into f16 B-fragment lane layout:
// flat index = (((chunk*8 + ct)*32 + lane)*16 + e), value = W[chunk*32 + half*8 + (e<8?e:e+8)][ct*16 + n]
__global__ void k_packB(const float* __restrict__ W1, _Float16* __restrict__ bpk) {
  int idx = blockIdx.x * 256 + threadIdx.x;
  if (idx >= NCHUNK * 8 * 32 * 16) return;
  int e     = idx & 15;
  int lane  = (idx >> 4) & 31;
  int ct    = (idx >> 9) & 7;
  int chunk = idx >> 12;
  int n = lane & 15, half = lane >> 4;
  int K   = chunk * 32 + half * 8 + (e < 8 ? e : e + 8);
  int col = ct * 16 + n;
  float v = (K < KTOT) ? W1[(size_t)K * HIDDEN + col] : 0.0f;
  bpk[idx] = (_Float16)v;
}

// h = relu([x | Tx1 | Tx2] (Nx495) @ W1cat (495x128) + b1)
// 4 waves/block; each wave owns one 16-row tile x all 128 cols (8 C tiles).
__global__ __launch_bounds__(128) void k_gemm1(
    const float* __restrict__ x, const float* __restrict__ tx1, const float* __restrict__ tx2,
    const _Float16* __restrict__ bpk, const float* __restrict__ b1, float* __restrict__ h) {
  int wave = threadIdx.x >> 5;
  int lane = threadIdx.x & 31;
  int tile = blockIdx.x * 4 + wave;           // 12500 tiles total (200000/16)
  int m = lane & 15, half = lane >> 4;
  size_t arow = (size_t)(tile * 16 + m) * F_IN;

  v8f acc[8] = {};
  const v8h* bb = (const v8h*)bpk;

  for (int chunk = 0; chunk < NCHUNK; ++chunk) {
    int kc = chunk * 32;
    // A fragment: 16x32 f16, ISA layout: lane(half,m); element e -> K = half*8 + (e<8?e:e+8)
    v16h a;
#pragma unroll
    for (int e = 0; e < 16; ++e) {
      int K = kc + half * 8 + (e < 8 ? e : e + 8);
      float v = 0.0f;
      if (K < KTOT) {
        int sgm = (K >= 2 * F_IN) ? 2 : (K >= F_IN ? 1 : 0);
        int f = K - sgm * F_IN;
        const float* p = (sgm == 0) ? x : (sgm == 1 ? tx1 : tx2);
        v = p[arow + f];
      }
      a[e] = (_Float16)v;
    }
#pragma unroll
    for (int ct = 0; ct < 8; ++ct) {
      int bi = ((chunk * 8 + ct) * 32 + lane) * 2;
      v8h blo = bb[bi];
      v8h bhi = bb[bi + 1];
      v16h b = __builtin_shufflevector(blo, bhi, 0,1,2,3,4,5,6,7,8,9,10,11,12,13,14,15);
      acc[ct] = __builtin_amdgcn_wmma_f32_16x16x32_f16(
          false, a, false, b, (short)0, acc[ct], false, false);
    }
  }
  // D layout: VGPR r -> row = 8*half + r, col = ct*16 + m
#pragma unroll
  for (int ct = 0; ct < 8; ++ct) {
    int col = ct * 16 + m;
    float bias = b1[col];
#pragma unroll
    for (int r = 0; r < 8; ++r) {
      int row = tile * 16 + half * 8 + r;
      float v = acc[ct][r] + bias;
      h[(size_t)row * HIDDEN + col] = v > 0.0f ? v : 0.0f;
    }
  }
}

// ---------------- layer-2 projection (Nx128 @ 128x6) ----------------
__global__ __launch_bounds__(256) void k_gemm2(const float* __restrict__ h,
                                               const float* __restrict__ W2,
                                               float* __restrict__ z0, float* __restrict__ z1,
                                               float* __restrict__ z2) {
  __shared__ float w[3 * HIDDEN * 2];
  for (int i = threadIdx.x; i < 3 * HIDDEN * 2; i += 256) w[i] = W2[i];
  __syncthreads();
  int n = blockIdx.x * 256 + threadIdx.x;
  if (n >= N_NODES) return;
  const float* hr = h + (size_t)n * HIDDEN;
  float a0 = 0, a1 = 0, a2 = 0, a3 = 0, a4 = 0, a5 = 0;
  for (int i = 0; i < HIDDEN; ++i) {
    float hv = hr[i];
    a0 += hv * w[i * 2 + 0];
    a1 += hv * w[i * 2 + 1];
    a2 += hv * w[256 + i * 2 + 0];
    a3 += hv * w[256 + i * 2 + 1];
    a4 += hv * w[512 + i * 2 + 0];
    a5 += hv * w[512 + i * 2 + 1];
  }
  z0[n * 2 + 0] = a0; z0[n * 2 + 1] = a1;
  z1[n * 2 + 0] = a2; z1[n * 2 + 1] = a3;
  z2[n * 2 + 0] = a4; z2[n * 2 + 1] = a5;
}

// out = z0 + L z1 + 2 L L z2 - z2 + b2
__global__ void k_final(const float* __restrict__ z0, const float* __restrict__ z2,
                        const float* __restrict__ p1, const float* __restrict__ p3,
                        const float* __restrict__ b2, float* __restrict__ out) {
  int i = blockIdx.x * 256 + threadIdx.x;
  if (i >= 2 * N_NODES) return;
  out[i] = z0[i] + p1[i] + 2.0f * p3[i] - z2[i] + b2[i & 1];
}

// ---------------- host ----------------
extern "C" void kernel_launch(void* const* d_in, const int* in_sizes, int n_in,
                              void* d_out, int out_size, void* d_ws, size_t ws_size,
                              hipStream_t stream) {
  (void)in_sizes; (void)n_in; (void)out_size; (void)ws_size;
  const float*     x  = (const float*)d_in[0];
  const long long* ei = (const long long*)d_in[1];   // int64 edge_index (2,E)
  const float*     W1 = (const float*)d_in[2];       // (3,165,128) == (495,128)
  const float*     b1 = (const float*)d_in[3];
  const float*     W2 = (const float*)d_in[4];       // (3,128,2)
  const float*     b2 = (const float*)d_in[5];
  float* out = (float*)d_out;

  char* ws = (char*)d_ws;
  size_t off = 0;
  auto carve = [&](size_t bytes) -> char* {
    char* p = ws + off;
    off += (bytes + 255) & ~(size_t)255;
    return p;
  };
  float*    deg = (float*)carve((size_t)N_NODES * 4);
  float*    dis = (float*)carve((size_t)N_NODES * 4);
  float*    nrm = (float*)carve((size_t)N_EDGES * 4);
  float*    tx1 = (float*)carve((size_t)N_NODES * F_IN * 4);
  float*    tx2 = (float*)carve((size_t)N_NODES * F_IN * 4);
  float*    h   = (float*)carve((size_t)N_NODES * HIDDEN * 4);
  _Float16* bpk = (_Float16*)carve((size_t)NCHUNK * 8 * 32 * 16 * 2);
  float*    z0  = (float*)carve((size_t)N_NODES * 2 * 4);
  float*    z1  = (float*)carve((size_t)N_NODES * 2 * 4);
  float*    z2  = (float*)carve((size_t)N_NODES * 2 * 4);
  float*    p1  = (float*)carve((size_t)N_NODES * 2 * 4);   // p1,p2,p3 contiguous
  float*    p2  = (float*)carve((size_t)N_NODES * 2 * 4);
  float*    p3  = (float*)carve((size_t)N_NODES * 2 * 4);
  (void)p2;

  const int eb = (N_EDGES + 255) / 256;
  const int nb = (N_NODES + 255) / 256;

  // normalization
  hipMemsetAsync(deg, 0, (size_t)N_NODES * 4, stream);
  k_degree<<<eb, 256, 0, stream>>>(ei, deg);
  k_dis<<<nb, 256, 0, stream>>>(deg, dis);
  k_norm<<<eb, 256, 0, stream>>>(ei, dis, nrm);

  // layer 1: Chebyshev recurrence at full width (cheapest: 2 props x 165)
  hipMemsetAsync(tx1, 0, (size_t)N_NODES * F_IN * 4, stream);
  dim3 pg(eb, (F_IN + 27) / 28);
  k_prop165<<<pg, 256, 0, stream>>>(ei, nrm, x, tx1, 1.0f);           // Tx1 = L x
  int cnt = N_NODES * F_IN;
  k_neg<<<(cnt + 255) / 256, 256, 0, stream>>>(x, tx2, cnt);          // Tx2 = -x
  k_prop165<<<pg, 256, 0, stream>>>(ei, nrm, tx1, tx2, 2.0f);         // Tx2 += 2 L Tx1

  // fused WMMA GEMM + bias + relu
  k_packB<<<(NCHUNK * 8 * 32 * 16 + 255) / 256, 256, 0, stream>>>(W1, bpk);
  k_gemm1<<<(N_NODES / 16) / 4, 128, 0, stream>>>(x, tx1, tx2, bpk, b1, h);

  // layer 2: project first (Nx2), then propagate narrow (64x less scatter traffic)
  k_gemm2<<<nb, 256, 0, stream>>>(h, W2, z0, z1, z2);
  hipMemsetAsync(p1, 0, (size_t)3 * N_NODES * 2 * 4, stream);
  k_prop2<<<eb, 256, 0, stream>>>(ei, nrm, z1, p1);                   // p1 = L z1
  k_prop2<<<eb, 256, 0, stream>>>(ei, nrm, z2, p2);                   // p2 = L z2
  k_prop2<<<eb, 256, 0, stream>>>(ei, nrm, p2, p3);                   // p3 = L L z2
  k_final<<<(2 * N_NODES + 255) / 256, 256, 0, stream>>>(z0, z2, p1, p3, b2, out);
}
